// SpatialAttentionBlock_17557826306713
// MI455X (gfx1250) — matrix operand order repaired and checked by reading, not compiled, not run on hardware
//
#include <hip/hip_runtime.h>

typedef __attribute__((ext_vector_type(16))) _Float16     v16h;
typedef __attribute__((ext_vector_type(8)))  float        v8f;
typedef __attribute__((ext_vector_type(8)))  unsigned int v8u;

#define B_   32
#define H_   56
#define W_   56
#define C_   256
#define NPIX (B_ * H_ * W_)
#define TH   16
#define TW   16
#define HALO 22          // TH + 6
#define HPAD 24          // halo row padded to 24 pixels (48 halves), zero-filled

union CvtAB { v8u u; v16h h; };

// ---------------------------------------------------------------------------
// Kernel 1: per-pixel channel mean + max. One wave32 per pixel, coalesced
// float4 loads, shfl_xor butterfly reduction. pool layout: [pix][2] = {avg,max}
// ---------------------------------------------------------------------------
__global__ __launch_bounds__(256) void sab_pool_kernel(const float* __restrict__ x,
                                                       float* __restrict__ pool)
{
    const int wave = threadIdx.x >> 5;
    const int lane = threadIdx.x & 31;
    const int pix  = blockIdx.x * 8 + wave;

    const float4* xb = (const float4*)(x + (size_t)pix * C_);
    float4 a = xb[lane];        // channels  lane*4 .. lane*4+3
    float4 b = xb[lane + 32];   // channels 128+lane*4 ..

    float s = a.x + a.y + a.z + a.w + b.x + b.y + b.z + b.w;
    float m = fmaxf(fmaxf(fmaxf(a.x, a.y), fmaxf(a.z, a.w)),
                    fmaxf(fmaxf(b.x, b.y), fmaxf(b.z, b.w)));

#pragma unroll
    for (int off = 16; off > 0; off >>= 1) {
        s += __shfl_xor(s, off, 32);
        m  = fmaxf(m, __shfl_xor(m, off, 32));
    }

    if (lane == 0) {
        pool[(size_t)pix * 2 + 0] = s * (1.0f / C_);
        pool[(size_t)pix * 2 + 1] = m;
    }
}

// ---------------------------------------------------------------------------
// Kernel 2: 7x7 conv (2->1 ch) on pooled map via WMMA, sigmoid, apply to x.
// Block = 16x16 spatial tile of one image; 8 waves, wave w does rows 2w,2w+1.
// Row's 16 pixels = M of v_wmma_f32_16x16x32_f16; K = 14 (7 dx * 2 ch) padded
// to 32 with zeros; accumulated over 7 dy taps; weights replicated across N.
// A/B operands built branch-free from pre-converted fp16 LDS buffers.
// ---------------------------------------------------------------------------
__global__ __launch_bounds__(256) void sab_conv_apply_kernel(const float* __restrict__ x,
                                                             const float* __restrict__ w,
                                                             const float* __restrict__ bias_p,
                                                             const float* __restrict__ pool,
                                                             float* __restrict__ out)
{
    __shared__ __align__(16) _Float16 hpool[HALO * HPAD * 2]; // [hy][hx*2+c], zero-padded
    __shared__ __align__(32) _Float16 hw[7 * 16];             // [dy][k], zero past k=13
    __shared__ float attn_s[TH * TW];

    const int bimg = blockIdx.z;
    const int ty0  = blockIdx.y * TH;
    const int tx0  = blockIdx.x * TW;
    const int tid  = threadIdx.x;

    // ---- load pooled halo, convert to fp16, zero-pad (spatial + row pad) ----
    for (int idx = tid; idx < HALO * HPAD * 2; idx += 256) {
        int hy  = idx / (HPAD * 2);
        int off = idx - hy * (HPAD * 2);
        int hx  = off >> 1;
        int c   = off & 1;
        int gy  = ty0 - 3 + hy;
        int gx  = tx0 - 3 + hx;
        float v = 0.0f;
        if (hx < HALO && gy >= 0 && gy < H_ && gx >= 0 && gx < W_)
            v = pool[((((size_t)bimg) * H_ + gy) * W_ + gx) * 2 + c];
        hpool[idx] = (_Float16)v;
    }
    // ---- weights -> fp16 LDS, k in [0,16) zero-padded past 13 ----
    if (tid < 7 * 16) {
        int k = tid & 15;
        float v = (k < 14) ? w[(tid >> 4) * 14 + k] : 0.0f;
        hw[tid] = (_Float16)v;
    }
    __syncthreads();

    const int  wave = tid >> 5;
    const int  lane = tid & 31;
    const int  m    = lane & 31 & 15;  // pixel (M) index within the row
    const bool lo   = (lane < 16);
    const float bias = bias_p[0];
    const int  r0   = wave * 2;

    v8f acc0 = {};
    v8f acc1 = {};

#pragma unroll
    for (int dy = 0; dy < 7; ++dy) {
        // B operand: lanes 0-15 carry K=e (e<8); lanes 16-31 carry K=8+e (e<8); e>=8 (K>=16) -> 0
        CvtAB b;
        {
            v8u wu = *(const v8u*)&hw[dy * 16];   // 32B aligned LDS load
#pragma unroll
            for (int i = 0; i < 8; ++i) b.u[i] = (i < 4) ? wu[i + (lo ? 0 : 4)] : 0u;
        }

        // A operands: lanes 0-15 need halves prow[0..7] (K=0..7);
        // lanes 16-31 need prow[8..13] then zeros (K=8..13,14,15); K>=16 slots zero.
#pragma unroll
        for (int rr = 0; rr < 2; ++rr) {
            const int r = r0 + rr;
            const unsigned* ap =
                (const unsigned*)&hpool[(r + dy) * (HPAD * 2) + m * 2] + (lo ? 0 : 4);
            unsigned d0 = ap[0];
            unsigned d1 = ap[1];
            unsigned d2 = ap[2];
            unsigned d3 = ap[3];            // unconditional: always in-bounds (row pad)
            d3 = lo ? d3 : 0u;              // zero K=14,15 pad for high lane group
            CvtAB a;
            a.u[0] = d0; a.u[1] = d1; a.u[2] = d2; a.u[3] = d3;
            a.u[4] = 0u; a.u[5] = 0u; a.u[6] = 0u; a.u[7] = 0u;
            if (rr == 0)
                acc0 = __builtin_amdgcn_wmma_f32_16x16x32_f16(
                    false, a.h, false, b.h, (short)0, acc0, false, false);
            else
                acc1 = __builtin_amdgcn_wmma_f32_16x16x32_f16(
                    false, a.h, false, b.h, (short)0, acc1, false, false);
        }
    }

    // D layout: lanes 0-15 hold M=v in acc[v]; lanes 16-31 hold M=v+8.
    // All N columns equal -> lanes with N==0 (lane 0 / lane 16) extract.
    if (lane == 0 || lane == 16) {
        const int pbase = lo ? 0 : 8;
#pragma unroll
        for (int v = 0; v < 8; ++v) {
            float t0 = acc0[v] + bias;
            float t1 = acc1[v] + bias;
            attn_s[r0 * TW + pbase + v]       = 1.0f / (1.0f + __expf(-t0));
            attn_s[(r0 + 1) * TW + pbase + v] = 1.0f / (1.0f + __expf(-t1));
        }
    }
    __syncthreads();

    // ---- apply: out = x * attn, float4 coalesced, 4 pixels x 64 lanes/iter --
    const int sub   = tid >> 6;          // 0..3  (pixel subgroup)
    const int chan4 = (tid & 63) * 4;    // channel offset
    const size_t tile_pix = ((size_t)bimg * H_ + ty0) * W_ + tx0;

    if (tx0 + TW <= W_ && ty0 + TH <= H_) {
        // interior tile: no per-pixel bounds checks (block-uniform branch)
#pragma unroll 4
        for (int i = 0; i < 64; ++i) {
            int p  = i * 4 + sub;            // tile-local pixel 0..255
            int py = p >> 4;
            int px = p & 15;
            float a = attn_s[p];
            size_t base = (tile_pix + (size_t)py * W_ + px) * (size_t)C_ + chan4;
            float4 xv = *(const float4*)(x + base);
            float4 ov;
            ov.x = xv.x * a; ov.y = xv.y * a; ov.z = xv.z * a; ov.w = xv.w * a;
            *(float4*)(out + base) = ov;
        }
    } else {
        // edge tile: clip to image
#pragma unroll 4
        for (int i = 0; i < 64; ++i) {
            int p  = i * 4 + sub;
            int py = p >> 4;
            int px = p & 15;
            int gy = ty0 + py;
            int gx = tx0 + px;
            if (gy < H_ && gx < W_) {
                float a = attn_s[p];
                size_t base = (tile_pix + (size_t)py * W_ + px) * (size_t)C_ + chan4;
                float4 xv = *(const float4*)(x + base);
                float4 ov;
                ov.x = xv.x * a; ov.y = xv.y * a; ov.z = xv.z * a; ov.w = xv.w * a;
                *(float4*)(out + base) = ov;
            }
        }
    }
}

// ---------------------------------------------------------------------------
extern "C" void kernel_launch(void* const* d_in, const int* in_sizes, int n_in,
                              void* d_out, int out_size, void* d_ws, size_t ws_size,
                              hipStream_t stream)
{
    (void)in_sizes; (void)n_in; (void)out_size; (void)ws_size;

    const float* x      = (const float*)d_in[0];
    const float* conv_w = (const float*)d_in[1];   // [7][7][2][1] HWIO
    const float* conv_b = (const float*)d_in[2];   // [1]
    float*       out    = (float*)d_out;
    float*       pool   = (float*)d_ws;            // NPIX * 2 floats

    // Kernel 1: 8 pixels per 256-thread block
    sab_pool_kernel<<<NPIX / 8, 256, 0, stream>>>(x, pool);

    // Kernel 2: 4 x 4 tiles of 16x16 per image, 32 images
    dim3 grid((W_ + TW - 1) / TW, (H_ + TH - 1) / TH, B_);
    sab_conv_apply_kernel<<<grid, 256, 0, stream>>>(x, conv_w, conv_b, pool, out);
}
